// GNN_5592047419839
// MI455X (gfx1250) — matrix operand order repaired and checked by reading, not compiled, not run on hardware
//
#include <hip/hip_runtime.h>
#include <math.h>

#define N_NODES 100000
#define F 128
#define EDGES 1600000

typedef __attribute__((ext_vector_type(16))) __bf16         v16bf;
typedef __attribute__((ext_vector_type(8)))  float          v8f;
typedef __attribute__((ext_vector_type(16))) unsigned short v16u;
typedef __attribute__((ext_vector_type(8)))  unsigned short v8u;

// fp32 -> bf16 with round-to-nearest-even
__device__ __forceinline__ unsigned short f2bf(float f) {
  unsigned u = __builtin_bit_cast(unsigned, f);
  unsigned r = u + 0x7FFFu + ((u >> 16) & 1u);
  return (unsigned short)(r >> 16);
}

__global__ void zero_kernel(float* __restrict__ p, int n) {
  int i = blockIdx.x * blockDim.x + threadIdx.x;
  if (i < n) p[i] = 0.0f;
}

__global__ void degree_kernel(const int* __restrict__ dst, float* __restrict__ deg, int e) {
  int i = blockIdx.x * blockDim.x + threadIdx.x;
  if (i < e) atomicAdd(&deg[dst[i]], 1.0f);
}

__global__ void dinv_kernel(const float* __restrict__ deg, float* __restrict__ dinv, int n) {
  int i = blockIdx.x * blockDim.x + threadIdx.x;
  if (i < n) dinv[i] = rsqrtf(deg[i] + 1.0f);  // +1 self-loop
}

// H[M,128] = X[M,128] @ W[128,128], bf16 WMMA, fp32 accumulate.
// 256 threads = 8 waves; block handles 16 rows; wave w owns cols [16w,16w+16).
__global__ __launch_bounds__(256) void gemm_wmma_kernel(const float* __restrict__ X,
                                                        const float* __restrict__ W,
                                                        float* __restrict__ H) {
  __shared__ unsigned short As[16 * 128];    // row-major  [m][k], bf16
  __shared__ unsigned short Bs[128 * 128];   // col-major  [n][k], bf16
  const int t = threadIdx.x;
  const int row_base = blockIdx.x * 16;

  // stage W (row-major W[k][n] in memory) transposed into Bs[n][k]
  for (int i = t; i < 128 * 128; i += 256) {
    int k = i >> 7, n = i & 127;
    Bs[n * 128 + k] = f2bf(W[k * 128 + n]);
  }
  // stage 16-row A strip
  for (int i = t; i < 16 * 128; i += 256) {
    int m = i >> 7, k = i & 127;
    As[i] = f2bf(X[(size_t)(row_base + m) * 128 + k]);
  }
  __syncthreads();

  const int wave = t >> 5;
  const int lane = t & 31;
  const int half = lane >> 4;   // 0: lanes 0-15, 1: lanes 16-31
  const int m16  = lane & 15;
  const int col  = wave * 16 + m16;   // B column owned by this lane

  v8f c = {};
#pragma unroll
  for (int kb = 0; kb < 128; kb += 32) {
    // A lane layout (ISA 7.12.2): VGPR0-3 -> K = kb+8*half+{0..7}; VGPR4-7 -> +16
    v8u alo = *(const v8u*)&As[m16 * 128 + kb + 8 * half];
    v8u ahi = *(const v8u*)&As[m16 * 128 + kb + 16 + 8 * half];
    // B lane layout: column N = m16, K = kb+8*half+{0..7}, then +16 (same as A, col-major LDS)
    v8u blo = *(const v8u*)&Bs[col * 128 + kb + 8 * half];
    v8u bhi = *(const v8u*)&Bs[col * 128 + kb + 16 + 8 * half];
    v16u au, bu;
#pragma unroll
    for (int j = 0; j < 8; ++j) {
      au[j] = alo[j]; au[j + 8] = ahi[j];
      bu[j] = blo[j]; bu[j + 8] = bhi[j];
    }
    v16bf a = __builtin_bit_cast(v16bf, au);
    v16bf b = __builtin_bit_cast(v16bf, bu);
    // (neg_a, A, neg_b, B, c_mod, C, reuse_a, reuse_b)
    c = __builtin_amdgcn_wmma_f32_16x16x32_bf16(false, a, false, b, (short)0, c, false, false);
  }

  // C/D layout: VGPR r -> M = r + 8*half, N = m16
#pragma unroll
  for (int r = 0; r < 8; ++r)
    H[(size_t)(row_base + half * 8 + r) * 128 + wave * 16 + m16] = c[r];
}

// One wave per edge; lane-contiguous features -> coalesced gather + atomic stream.
__global__ void scatter_kernel(const float* __restrict__ h,
                               const int* __restrict__ src,
                               const int* __restrict__ dst,
                               const float* __restrict__ dinv,
                               float* __restrict__ agg, int e) {
  int gid  = blockIdx.x * blockDim.x + threadIdx.x;
  int edge = gid >> 5;
  int lane = gid & 31;
  if (edge >= e) return;
  int s = src[edge], d = dst[edge];
  float norm = dinv[s] * dinv[d];
  const float* hr = h + (size_t)s * F;
  float* ar = agg + (size_t)d * F;
#pragma unroll
  for (int k = 0; k < 4; ++k) {
    int f = lane + 32 * k;
    atomicAdd(&ar[f], hr[f] * norm);
  }
}

// z = relu(agg + h*dinv^2 + bias)
__global__ void finalize_kernel(const float* __restrict__ agg,
                                const float* __restrict__ h,
                                const float* __restrict__ dinv,
                                const float* __restrict__ bias,
                                float* __restrict__ z, int total) {
  int i = blockIdx.x * blockDim.x + threadIdx.x;
  if (i >= total) return;
  int node = i >> 7, f = i & 127;
  float di = dinv[node];
  float v = agg[i] + h[i] * di * di + bias[f];
  z[i] = v > 0.0f ? v : 0.0f;
}

// wave per node: two 128-length dots, wave32 shfl_xor reduction, log-softmax
__global__ void fc_logsoftmax_kernel(const float* __restrict__ z,
                                     const float* __restrict__ Wfc,
                                     const float* __restrict__ bfc,
                                     float* __restrict__ out, int n) {
  int gid  = blockIdx.x * blockDim.x + threadIdx.x;
  int node = gid >> 5;
  int lane = gid & 31;
  if (node >= n) return;
  const float* zr = z + (size_t)node * F;
  float a0 = 0.0f, a1 = 0.0f;
#pragma unroll
  for (int k = 0; k < 4; ++k) {
    int f = lane + 32 * k;
    float v = zr[f];
    a0 += v * Wfc[f * 2 + 0];
    a1 += v * Wfc[f * 2 + 1];
  }
#pragma unroll
  for (int m = 16; m >= 1; m >>= 1) {
    a0 += __shfl_xor(a0, m, 32);
    a1 += __shfl_xor(a1, m, 32);
  }
  if (lane == 0) {
    float l0 = a0 + bfc[0], l1 = a1 + bfc[1];
    float mx = fmaxf(l0, l1);
    float lse = mx + logf(expf(l0 - mx) + expf(l1 - mx));
    out[node * 2 + 0] = l0 - lse;
    out[node * 2 + 1] = l1 - lse;
  }
}

extern "C" void kernel_launch(void* const* d_in, const int* in_sizes, int n_in,
                              void* d_out, int out_size, void* d_ws, size_t ws_size,
                              hipStream_t stream) {
  const float* x   = (const float*)d_in[0];
  const int*   ei  = (const int*)d_in[1];   // edge_index [2, E]
  const float* W1  = (const float*)d_in[2];
  const float* b1  = (const float*)d_in[3];
  const float* W2  = (const float*)d_in[4];
  const float* b2  = (const float*)d_in[5];
  const float* Wfc = (const float*)d_in[6];
  const float* bfc = (const float*)d_in[7];
  float* out = (float*)d_out;

  const int* src = ei;
  const int* dst = ei + EDGES;

  // workspace layout (all fp32): deg | dinv | h | agg | z
  float* deg    = (float*)d_ws;
  float* dinv   = deg + N_NODES;
  float* hbuf   = dinv + N_NODES;
  float* aggbuf = hbuf + (size_t)N_NODES * F;
  float* zbuf   = aggbuf + (size_t)N_NODES * F;

  const dim3 B(256);
  const int nodeBlocks = (N_NODES + 255) / 256;
  const int featTotal  = N_NODES * F;
  const int featBlocks = (featTotal + 255) / 256;
  const int edgeBlocks = (EDGES + 255) / 256;
  const int scatBlocks = (int)(((long long)EDGES * 32 + 255) / 256);
  const int gemmBlocks = N_NODES / 16;  // 100000 % 16 == 0
  const int fcBlocks   = (int)(((long long)N_NODES * 32 + 255) / 256);

  // degree + normalization
  zero_kernel<<<nodeBlocks, B, 0, stream>>>(deg, N_NODES);
  degree_kernel<<<edgeBlocks, B, 0, stream>>>(dst, deg, EDGES);
  dinv_kernel<<<nodeBlocks, B, 0, stream>>>(deg, dinv, N_NODES);

  // layer 1
  gemm_wmma_kernel<<<gemmBlocks, B, 0, stream>>>(x, W1, hbuf);
  zero_kernel<<<featBlocks, B, 0, stream>>>(aggbuf, featTotal);
  scatter_kernel<<<scatBlocks, B, 0, stream>>>(hbuf, src, dst, dinv, aggbuf, EDGES);
  finalize_kernel<<<featBlocks, B, 0, stream>>>(aggbuf, hbuf, dinv, b1, zbuf, featTotal);

  // layer 2
  gemm_wmma_kernel<<<gemmBlocks, B, 0, stream>>>(zbuf, W2, hbuf);
  zero_kernel<<<featBlocks, B, 0, stream>>>(aggbuf, featTotal);
  scatter_kernel<<<scatBlocks, B, 0, stream>>>(hbuf, src, dst, dinv, aggbuf, EDGES);
  finalize_kernel<<<featBlocks, B, 0, stream>>>(aggbuf, hbuf, dinv, b2, zbuf, featTotal);

  // head
  fc_logsoftmax_kernel<<<fcBlocks, B, 0, stream>>>(zbuf, Wfc, bfc, out, N_NODES);
}
